// SyntaxesAttention_45483703665004
// MI455X (gfx1250) — compile-verified, matrix operand order checked by reading, not run on hardware
//
#include <hip/hip_runtime.h>
#include <hip/hip_bf16.h>
#include <math.h>

typedef __bf16 bf16;
typedef __attribute__((ext_vector_type(16))) __bf16 v16bf;
typedef __attribute__((ext_vector_type(8)))  __bf16 v8bf;
typedef __attribute__((ext_vector_type(4)))  __bf16 v4bf;
typedef __attribute__((ext_vector_type(8)))  float  v8f;

#define BB 4
#define SS 4096
#define HH 1024
#define NHEAD 16
#define HD 64
#define KC 64
#define ATT_SCALE 0.125f   /* 1/sqrt(64) */

// GEMM tile
#define TM 128
#define TN 64
#define TK 32
#define LDS_K (TK + 8)

// ---------------------------------------------------------------------------
// Async global->LDS copy (16B per lane), tracked by ASYNCcnt — keeps staging
// off the LOADcnt/DScnt counters the WMMA ds_load stream waits on.
// Generic LDS pointer: low 32 bits of the flat address are the DS offset.
// ---------------------------------------------------------------------------
__device__ __forceinline__ void async_cp16(void* lds_ptr, const void* gptr) {
  unsigned lo = (unsigned)(unsigned long long)lds_ptr;
  asm volatile("global_load_async_to_lds_b128 %0, %1, off"
               :: "v"(lo), "v"((unsigned long long)gptr)
               : "memory");
}
__device__ __forceinline__ void async_wait0() {
#if __has_builtin(__builtin_amdgcn_s_wait_asynccnt)
  __builtin_amdgcn_s_wait_asynccnt(0);
#else
  asm volatile("s_wait_asynccnt 0x0" ::: "memory");
#endif
}

// ---------------------------------------------------------------------------
// WMMA fragment helpers (ISA 7.12.2): A fragment = two contiguous 8xbf16
// K-chunks per lane (two 16B LDS loads + shuffle); B stored [n][k] in LDS
// loads with the identical pattern.
// ---------------------------------------------------------------------------
__device__ __forceinline__ v16bf frag_ld(const bf16* rowp, int half) {
  const v8bf lo = *reinterpret_cast<const v8bf*>(rowp + half * 8);
  const v8bf hi = *reinterpret_cast<const v8bf*>(rowp + 16 + half * 8);
  return __builtin_shufflevector(lo, hi, 0,1,2,3,4,5,6,7,8,9,10,11,12,13,14,15);
}

__device__ __forceinline__ v8f wmma_bf16(v16bf a, v16bf b, v8f c) {
  return __builtin_amdgcn_wmma_f32_16x16x32_bf16(false, a, false, b,
                                                 (short)0, c, false, false);
}

// ---------------------------------------------------------------------------
// Templated WMMA GEMM:  C = op(A) * op(B) (+ Cadd), batched via gridDim.z.
//   AM: 0 = f32 [M,Kd] row-major     (float4 reg staging + convert)
//       1 = f32 [Kd,M] (A^T access)  (coalesced scalar staging, clamped M)
//       2 = bf16 [M,Kd] row-major    (ASYNC global->LDS, no VGPR round trip)
//   BMO: 0 = bf16 [N,Kd] (pre-transposed weight, ASYNC global->LDS)
//        1 = f32  [Kd,N] (coalesced scalar transpose staging)
//   CM: 0 = f32 out (+optional Cadd), 1 = bf16 out
// Double-buffered LDS; tile t+1 prefetch (async or reg) overlaps tile t WMMAs.
// Branch-free staging: rows >= M are clamped duplicates, never stored.
// ---------------------------------------------------------------------------
template<int AM, int BMO, int CM>
__global__ __launch_bounds__(256)
void gemm_wmma(const void* __restrict__ Av, const void* __restrict__ Bv,
               void* __restrict__ Cv, const float* __restrict__ Cadd,
               int M, int N, int Kd, int lda, int ldb, int ldc,
               long long sA, long long sB, long long sC) {
  __shared__ __align__(16) bf16 As[2][TM][LDS_K];
  __shared__ __align__(16) bf16 Bs[2][TN][LDS_K];

  constexpr bool USE_ASYNC = (AM == 2) || (BMO == 0);

  const float* Af = nullptr; const bf16* Ab = nullptr;
  const float* Bf = nullptr; const bf16* Bb = nullptr;
  if constexpr (AM == 2) Ab = (const bf16*)Av + (long long)blockIdx.z * sA;
  else                   Af = (const float*)Av + (long long)blockIdx.z * sA;
  if constexpr (BMO == 0) Bb = (const bf16*)Bv + (long long)blockIdx.z * sB;
  else                    Bf = (const float*)Bv + (long long)blockIdx.z * sB;

  const int m0 = blockIdx.y * TM;
  const int n0 = blockIdx.x * TN;
  const int tid  = threadIdx.x;
  const int lane = tid & 31;
  const int wave = tid >> 5;
  const int wm = wave & 3;
  const int wn = wave >> 2;
  const int half = (lane >> 4) & 1;
  const int l16  = lane & 15;

  // prefetch registers (unused ones eliminated per instantiation)
  float4 pa_f[4]; float pa_t[16];
  float  pb_f[8];

  // ---- A staging -----------------------------------------------------------
  auto fetchA = [&](int kk0, int buf) {
    if constexpr (AM == 2) {
      #pragma unroll
      for (int u = 0; u < 2; ++u) {
        int i = tid + u * 256;
        int r = i >> 2, c8 = (i & 3) * 8;
        int gm = m0 + r; gm = gm < M ? gm : M - 1;
        async_cp16(&As[buf][r][c8], Ab + (long long)gm * lda + kk0 + c8);
      }
    } else if constexpr (AM == 0) {
      #pragma unroll
      for (int u = 0; u < 4; ++u) {
        int i = tid + u * 256;
        int r = i >> 3, c4 = (i & 7) * 4;
        int gm = m0 + r; gm = gm < M ? gm : M - 1;
        pa_f[u] = *reinterpret_cast<const float4*>(Af + (long long)gm * lda + kk0 + c4);
      }
    } else {
      #pragma unroll
      for (int u = 0; u < 16; ++u) {
        int i = tid + u * 256;
        int r = i & (TM - 1), c = i >> 7;
        int gm = m0 + r; gm = gm < M ? gm : M - 1;
        pa_t[u] = Af[(long long)(kk0 + c) * lda + gm];
      }
    }
  };
  auto commitA = [&](int buf) {   // reg -> LDS (non-async modes only)
    if constexpr (AM == 0) {
      #pragma unroll
      for (int u = 0; u < 4; ++u) {
        int i = tid + u * 256;
        int r = i >> 3, c4 = (i & 7) * 4;
        v4bf o = { (bf16)pa_f[u].x, (bf16)pa_f[u].y, (bf16)pa_f[u].z, (bf16)pa_f[u].w };
        *reinterpret_cast<v4bf*>(&As[buf][r][c4]) = o;
      }
    } else if constexpr (AM == 1) {
      #pragma unroll
      for (int u = 0; u < 16; ++u) {
        int i = tid + u * 256;
        int r = i & (TM - 1), c = i >> 7;
        As[buf][r][c] = (bf16)pa_t[u];
      }
    }
  };

  // ---- B staging -----------------------------------------------------------
  auto fetchB = [&](int kk0, int buf) {
    if constexpr (BMO == 0) {
      int i = tid;
      int n = i >> 2, c8 = (i & 3) * 8;
      async_cp16(&Bs[buf][n][c8], Bb + (long long)(n0 + n) * ldb + kk0 + c8);
    } else {
      #pragma unroll
      for (int u = 0; u < 8; ++u) {
        int i = tid + u * 256;
        int n = i & (TN - 1), c = i >> 6;
        pb_f[u] = Bf[(long long)(kk0 + c) * ldb + (n0 + n)];
      }
    }
  };
  auto commitB = [&](int buf) {
    if constexpr (BMO == 1) {
      #pragma unroll
      for (int u = 0; u < 8; ++u) {
        int i = tid + u * 256;
        int n = i & (TN - 1), c = i >> 6;
        Bs[buf][n][c] = (bf16)pb_f[u];
      }
    }
  };

  v8f acc[2][2] = {};

  fetchA(0, 0); fetchB(0, 0);
  commitA(0);   commitB(0);
  if constexpr (USE_ASYNC) async_wait0();
  __syncthreads();

  const int nt = Kd / TK;
  for (int t = 0; t < nt; ++t) {
    const int cur = t & 1;
    const bool more = (t + 1) < nt;
    if (more) { fetchA((t + 1) * TK, cur ^ 1); fetchB((t + 1) * TK, cur ^ 1); }

    v16bf a0 = frag_ld(&As[cur][wm * 32 + l16][0], half);
    v16bf a1 = frag_ld(&As[cur][wm * 32 + 16 + l16][0], half);
    v16bf b0 = frag_ld(&Bs[cur][wn * 32 + l16][0], half);
    v16bf b1 = frag_ld(&Bs[cur][wn * 32 + 16 + l16][0], half);
    acc[0][0] = wmma_bf16(a0, b0, acc[0][0]);
    acc[0][1] = wmma_bf16(a0, b1, acc[0][1]);
    acc[1][0] = wmma_bf16(a1, b0, acc[1][0]);
    acc[1][1] = wmma_bf16(a1, b1, acc[1][1]);

    if (more) { commitA(cur ^ 1); commitB(cur ^ 1); }
    if constexpr (USE_ASYNC) async_wait0();
    __syncthreads();
  }

  #pragma unroll
  for (int i = 0; i < 2; ++i)
    #pragma unroll
    for (int j = 0; j < 2; ++j)
      #pragma unroll
      for (int r = 0; r < 8; ++r) {
        int gm = m0 + wm * 32 + i * 16 + r + half * 8;
        int gn = n0 + wn * 32 + j * 16 + l16;
        if (gm < M) {
          long long idx = (long long)blockIdx.z * sC + (long long)gm * ldc + gn;
          float vv = acc[i][j][r];
          if constexpr (CM == 0) {
            if (Cadd) vv += Cadd[(long long)gm * ldc + gn];
            ((float*)Cv)[idx] = vv;
          } else {
            ((bf16*)Cv)[idx] = (bf16)vv;
          }
        }
      }
}

// ---------------------------------------------------------------------------
// One-shot prep: f32 -> bf16 elementwise, and f32 [R,C] -> bf16 [C,R].
// ---------------------------------------------------------------------------
__global__ __launch_bounds__(256)
void f32_to_bf16(const float* __restrict__ in, bf16* __restrict__ outp, long long n) {
  long long i = ((long long)blockIdx.x * 256 + threadIdx.x) * 4;
  if (i < n) {
    float4 v = *reinterpret_cast<const float4*>(in + i);
    v4bf o = { (bf16)v.x, (bf16)v.y, (bf16)v.z, (bf16)v.w };
    *reinterpret_cast<v4bf*>(outp + i) = o;
  }
}

__global__ __launch_bounds__(256)
void transpose_to_bf16(const float* __restrict__ in, bf16* __restrict__ outp,
                       int R, int C) {
  __shared__ float tile[32][33];
  const int bx = blockIdx.x * 32;   // C
  const int by = blockIdx.y * 32;   // R
  const int tx = threadIdx.x & 31, ty = threadIdx.x >> 5;
  #pragma unroll
  for (int yy = ty; yy < 32; yy += 8)
    tile[yy][tx] = in[(long long)(by + yy) * C + bx + tx];
  __syncthreads();
  #pragma unroll
  for (int yy = ty; yy < 32; yy += 8)
    outp[(long long)(bx + yy) * R + by + tx] = (bf16)tile[tx][yy];
}

// ---------------------------------------------------------------------------
// Column softmax over the sequence axis: logits [B*S, Kc], one block/(b,k).
// ---------------------------------------------------------------------------
__global__ __launch_bounds__(256)
void col_softmax(float* __restrict__ logits, int S, int Kc) {
  const int b = blockIdx.y, k = blockIdx.x;
  float* col = logits + (long long)b * S * Kc + k;
  __shared__ float red[256];
  const int tid = threadIdx.x;

  float mx = -3.0e38f;
  for (int i = tid; i < S; i += 256) mx = fmaxf(mx, col[(long long)i * Kc]);
  red[tid] = mx; __syncthreads();
  for (int off = 128; off > 0; off >>= 1) {
    if (tid < off) red[tid] = fmaxf(red[tid], red[tid + off]);
    __syncthreads();
  }
  mx = red[0]; __syncthreads();

  float sum = 0.0f;
  for (int i = tid; i < S; i += 256) sum += __expf(col[(long long)i * Kc] - mx);
  red[tid] = sum; __syncthreads();
  for (int off = 128; off > 0; off >>= 1) {
    if (tid < off) red[tid] += red[tid + off];
    __syncthreads();
  }
  const float inv = 1.0f / red[0];

  for (int i = tid; i < S; i += 256) {
    float e = __expf(col[(long long)i * Kc] - mx);
    col[(long long)i * Kc] = e * inv;
  }
}

// ---------------------------------------------------------------------------
// Fused attention per (b, head, 64-row S-tile), all-bf16 I/O:
//   S1 = Q K^T -> progressive mask + softmax -> P ; O = P V.
// Q/K tiles staged via async global->LDS; V needs a transpose so it goes
// through VGPRs.  Scores never leave LDS.
// ---------------------------------------------------------------------------
__global__ __launch_bounds__(256)
void attn_fused(const bf16* __restrict__ q, const bf16* __restrict__ kk,
                const bf16* __restrict__ vv, bf16* __restrict__ out) {
  const int b  = blockIdx.z;
  const int nh = blockIdx.y;
  const int s0 = blockIdx.x * 64;

  __shared__ __align__(16) bf16 Qs[64][72];   // [s][d]
  __shared__ __align__(16) bf16 Ks[64][72];   // [j][d]
  __shared__ __align__(16) bf16 Vt[64][72];   // [d][j]
  __shared__ __align__(16) bf16 Ps[64][72];   // [s][j]
  __shared__ float satt[64][65];

  const int tid  = threadIdx.x;
  const int lane = tid & 31;
  const int wave = tid >> 5;
  const int wm = wave & 3;
  const int wn = wave >> 2;
  const int half = (lane >> 4) & 1;
  const int l16  = lane & 15;
  const int hcol = nh * HD;

  #pragma unroll
  for (int u = 0; u < 2; ++u) {
    int i = tid + u * 256;                    // 512 chunks of 8 bf16
    int r = i >> 3, c8 = (i & 7) * 8;
    async_cp16(&Qs[r][c8], q  + ((long long)(b * SS + s0 + r)) * HH + hcol + c8);
    async_cp16(&Ks[r][c8], kk + ((long long)(b * KC + r)) * HH + hcol + c8);
    v8bf vr = *reinterpret_cast<const v8bf*>(vv + ((long long)(b * KC + r)) * HH + hcol + c8);
    #pragma unroll
    for (int e = 0; e < 8; ++e) Vt[c8 + e][r] = vr[e];
  }
  async_wait0();
  __syncthreads();

  // ---- S1 = Q x K^T ----
  v8f acc[2] = {};
  #pragma unroll
  for (int kd = 0; kd < 64; kd += 32) {
    v16bf aq = frag_ld(&Qs[wm * 16 + l16][kd], half);
    v16bf b0 = frag_ld(&Ks[wn * 32 + l16][kd], half);
    v16bf b1 = frag_ld(&Ks[wn * 32 + 16 + l16][kd], half);
    acc[0] = wmma_bf16(aq, b0, acc[0]);
    acc[1] = wmma_bf16(aq, b1, acc[1]);
  }
  #pragma unroll
  for (int j = 0; j < 2; ++j)
    #pragma unroll
    for (int r = 0; r < 8; ++r)
      satt[wm * 16 + r + half * 8][wn * 32 + j * 16 + l16] = acc[j][r];
  __syncthreads();

  // ---- progressive mask + softmax (4 threads/row, wave32-safe shuffles) ----
  {
    const int row = tid >> 2;
    const int q4  = tid & 3;
    const int sg  = s0 + row;
    int allowed = (int)(((float)(sg + 1)) * (float)KC / (float)SS) + 1;
    if (allowed > KC) allowed = KC;

    float vals[16];
    float mx = -3.0e38f;
    #pragma unroll
    for (int t = 0; t < 16; ++t) {
      int j = q4 * 16 + t;
      float x = satt[row][j] * ATT_SCALE;
      if (j >= allowed) x = -1.0e9f;
      vals[t] = x;
      mx = fmaxf(mx, x);
    }
    mx = fmaxf(mx, __shfl_xor(mx, 1));
    mx = fmaxf(mx, __shfl_xor(mx, 2));
    float sum = 0.0f;
    #pragma unroll
    for (int t = 0; t < 16; ++t) { vals[t] = __expf(vals[t] - mx); sum += vals[t]; }
    sum += __shfl_xor(sum, 1);
    sum += __shfl_xor(sum, 2);
    const float inv = 1.0f / sum;
    #pragma unroll
    for (int t = 0; t < 16; ++t) Ps[row][q4 * 16 + t] = (bf16)(vals[t] * inv);
  }
  __syncthreads();

  // ---- O = P x V ----
  v8f oacc[2] = {};
  #pragma unroll
  for (int kj = 0; kj < 64; kj += 32) {
    v16bf ap = frag_ld(&Ps[wm * 16 + l16][kj], half);
    v16bf b0 = frag_ld(&Vt[wn * 32 + l16][kj], half);
    v16bf b1 = frag_ld(&Vt[wn * 32 + 16 + l16][kj], half);
    oacc[0] = wmma_bf16(ap, b0, oacc[0]);
    oacc[1] = wmma_bf16(ap, b1, oacc[1]);
  }
  #pragma unroll
  for (int j = 0; j < 2; ++j)
    #pragma unroll
    for (int r = 0; r < 8; ++r) {
      int m = wm * 16 + r + half * 8;
      int d = wn * 32 + j * 16 + l16;
      out[((long long)(b * SS + s0 + m)) * HH + hcol + d] = (bf16)oacc[j][r];
    }
}

// ---------------------------------------------------------------------------
extern "C" void kernel_launch(void* const* d_in, const int* in_sizes, int n_in,
                              void* d_out, int out_size, void* d_ws, size_t ws_size,
                              hipStream_t stream) {
  const float* inputs   = (const float*)d_in[0];
  const float* wq       = (const float*)d_in[1];
  const float* wk       = (const float*)d_in[2];
  const float* wv       = (const float*)d_in[3];
  const float* wo       = (const float*)d_in[4];
  const float* w_interp = (const float*)d_in[5];
  const float* pos_emb  = (const float*)d_in[6];
  float* out = (float*)d_out;

  const long long BS = (long long)BB * SS;                 // 16384
  float* ws = (float*)d_ws;
  long long o = 0;
  bf16*  xbf  = (bf16*)(ws + o);  o += BS * HH / 2;        // x   bf16 [BS,H]
  float* lg   = ws + o;           o += BS * KC;            // logits/iw f32
  float* comp = ws + o;           o += (long long)BB * KC * HH;   // f32 [B,K,H]
  bf16*  qb   = (bf16*)(ws + o);  o += BS * HH / 2;        // q   bf16
  bf16*  kkb  = (bf16*)(ws + o);  o += (long long)BB * KC * HH / 2;
  bf16*  vb   = (bf16*)(ws + o);  o += (long long)BB * KC * HH / 2;
  bf16*  ao   = (bf16*)(ws + o);  o += BS * HH / 2;        // attn out bf16
  bf16*  wqT  = (bf16*)(ws + o);  o += (long long)HH * HH / 2;
  bf16*  wkT  = (bf16*)(ws + o);  o += (long long)HH * HH / 2;
  bf16*  wvT  = (bf16*)(ws + o);  o += (long long)HH * HH / 2;
  bf16*  woT  = (bf16*)(ws + o);  o += (long long)HH * HH / 2;
  bf16*  wiT  = (bf16*)(ws + o);  o += (long long)KC * HH / 2;

  dim3 blk(256);

  // ---- prep: bf16 copy of x, transposed-bf16 weights (~us of traffic) ------
  f32_to_bf16<<<dim3((unsigned)(BS * HH / 4 / 256)), blk, 0, stream>>>(inputs, xbf, BS * HH);
  transpose_to_bf16<<<dim3(HH / 32, HH / 32), blk, 0, stream>>>(wq, wqT, HH, HH);
  transpose_to_bf16<<<dim3(HH / 32, HH / 32), blk, 0, stream>>>(wk, wkT, HH, HH);
  transpose_to_bf16<<<dim3(HH / 32, HH / 32), blk, 0, stream>>>(wv, wvT, HH, HH);
  transpose_to_bf16<<<dim3(HH / 32, HH / 32), blk, 0, stream>>>(wo, woT, HH, HH);
  transpose_to_bf16<<<dim3(KC / 32, HH / 32), blk, 0, stream>>>(w_interp, wiT, HH, KC);

  // 1) logits = x @ w_interp            [16384 x 64]  (argsort is a no-op)
  gemm_wmma<2, 0, 0><<<dim3(KC / TN, (unsigned)(BS / TM), 1), blk, 0, stream>>>(
      xbf, wiT, lg, nullptr, (int)BS, KC, HH, HH, HH, KC, 0, 0, 0);

  // 2) iw = softmax over S
  col_softmax<<<dim3(KC, BB), blk, 0, stream>>>(lg, SS, KC);

  // 3) compressed[b] = iw[b]^T @ x[b] + pos_emb     [64 x 1024], batched
  gemm_wmma<1, 1, 0><<<dim3(HH / TN, 1, BB), blk, 0, stream>>>(
      lg, inputs, comp, pos_emb, KC, HH, SS, KC, HH, HH,
      (long long)SS * KC, (long long)SS * HH, (long long)KC * HH);

  // 4) q = x @ wq -> bf16               [16384 x 1024]
  gemm_wmma<2, 0, 1><<<dim3(HH / TN, (unsigned)(BS / TM), 1), blk, 0, stream>>>(
      xbf, wqT, qb, nullptr, (int)BS, HH, HH, HH, HH, HH, 0, 0, 0);

  // 5) k/v = compressed @ wk/wv -> bf16 [256 x 1024]
  gemm_wmma<0, 0, 1><<<dim3(HH / TN, (BB * KC + TM - 1) / TM, 1), blk, 0, stream>>>(
      comp, wkT, kkb, nullptr, BB * KC, HH, HH, HH, HH, HH, 0, 0, 0);
  gemm_wmma<0, 0, 1><<<dim3(HH / TN, (BB * KC + TM - 1) / TM, 1), blk, 0, stream>>>(
      comp, wvT, vb, nullptr, BB * KC, HH, HH, HH, HH, HH, 0, 0, 0);

  // 6) fused masked attention (scores stay in LDS)
  attn_fused<<<dim3(SS / 64, NHEAD, BB), blk, 0, stream>>>(qb, kkb, vb, ao);

  // 7) out = attn @ wo                  [16384 x 1024] -> f32
  gemm_wmma<2, 0, 0><<<dim3(HH / TN, (unsigned)(BS / TM), 1), blk, 0, stream>>>(
      ao, woT, out, nullptr, (int)BS, HH, HH, HH, HH, HH, 0, 0, 0);
}
